// BiLSTM_CRF_MODIFY_PARALLEL_5617817223777
// MI455X (gfx1250) — matrix operand order repaired and checked by reading, not compiled, not run on hardware
//
#include <hip/hip_runtime.h>
#include <hip/hip_bf16.h>

// ---------------------------------------------------------------------------
// BiLSTM-CRF for MI455X (gfx1250, wave32).
//   K1: embedding gather + f32->bf16
//   K2: f32->bf16 weight convert (W_ih_f/b, W_hh_f/b)
//   K3: WMMA GEMM  Z = X @ W_ih^T + b_ih + b_hh   (v_wmma_f32_16x16x32_bf16)
//   K4: sequential LSTM recurrence via broadcast-A WMMA GEMV,
//       half of W_hh(bf16) LDS-resident, half streamed from L2
//   K5: output projection feats = [hf|hb] @ W_out^T + b_out
//   K6: Viterbi forward + backtrack (single block)
// ---------------------------------------------------------------------------

#define T_SEQ 4096
#define EMB   256
#define H2    256
#define G4    1024   // 4*H2 gate width
#define TAGS  6
#define TSTART 3
#define TSTOP  4
#define NEGV  (-10000.0f)

typedef __bf16 bf16_t;
typedef __attribute__((ext_vector_type(16))) __bf16 v16bf;
typedef __attribute__((ext_vector_type(8)))  __bf16 v8bf;
typedef __attribute__((ext_vector_type(8)))  float  v8f;

// ---------------- K1: embedding gather + bf16 convert ----------------------
__global__ __launch_bounds__(256) void gather_embed_bf16(
    const int* __restrict__ sent, const float* __restrict__ emb,
    bf16_t* __restrict__ X) {
  int idx = blockIdx.x * 256 + threadIdx.x;   // over T_SEQ * (EMB/4)
  int t  = idx >> 6;                          // EMB/4 = 64 quads per row
  int k4 = (idx & 63) << 2;
  if (t >= T_SEQ) return;
  float4 v = *(const float4*)(emb + (size_t)sent[t] * EMB + k4);
  bf16_t* o = X + (size_t)t * EMB + k4;
  o[0] = (bf16_t)v.x; o[1] = (bf16_t)v.y; o[2] = (bf16_t)v.z; o[3] = (bf16_t)v.w;
}

// ---------------- K2: f32 -> bf16 weight convert ---------------------------
__global__ __launch_bounds__(256) void cvt_bf16(
    const float* __restrict__ src, bf16_t* __restrict__ dst, int n) {
  int i = blockIdx.x * 256 + threadIdx.x;
  if (i < n) dst[i] = (bf16_t)src[i];
}

// ---------------- K3: WMMA GEMM  Z[t, j] = sum_k X[t,k] * Wih[j,k] + bias --
// One wave computes a 16x16 f32 tile of Z, K-loop over 256 in 8 WMMA steps.
// B-matrix = Wih^T: column N of B is row (j0+N) of row-major Wih ->
// a single contiguous 16-element bf16 load per lane (k = 16*hi + s, same k order as A).
__global__ __launch_bounds__(256) void gemm_z_wmma(
    const bf16_t* __restrict__ X,     // (T_SEQ, EMB) bf16
    const bf16_t* __restrict__ Wih,   // (G4, EMB)    bf16 row-major
    const float*  __restrict__ b_ih,  // (G4,)
    const float*  __restrict__ b_hh,  // (G4,)
    float* __restrict__ Z) {          // (T_SEQ, G4)  f32
  const int wave = threadIdx.x >> 5;            // 8 waves / block
  const int lane = threadIdx.x & 31;
  const int tile = blockIdx.x * 8 + wave;       // 256 * 64 tiles total
  const int tm = tile >> 6;                     // row tile (64 N-tiles)
  const int tn = tile & 63;
  const int t0 = tm * 16;
  const int j0 = tn * 16;
  const int l  = lane & 15;
  const int hi = lane >> 4;                     // lane group 0/1

  // A: lane group 0 -> K {0..15}; group 1 -> K {16..31} (per k0)
  const bf16_t* arow = X   + (size_t)(t0 + l) * EMB + hi * 16;
  // B: lane holds column j0+l, 16 contiguous K (group 1 offset +16)
  const bf16_t* brow = Wih + (size_t)(j0 + l) * EMB + hi * 16;

  v8f c = {};
#pragma unroll
  for (int k0 = 0; k0 < EMB; k0 += 32) {
    __builtin_prefetch(arow + k0 + 32, 0, 0);   // global_prefetch_b8
    __builtin_prefetch(brow + k0 + 32, 0, 0);
    v8bf a_lo = *(const v8bf*)(arow + k0);
    v8bf a_hi = *(const v8bf*)(arow + k0 + 8);
    v16bf a = __builtin_shufflevector(a_lo, a_hi,
                0, 1, 2, 3, 4, 5, 6, 7, 8, 9, 10, 11, 12, 13, 14, 15);
    v16bf b = *(const v16bf*)(brow + k0);
    c = __builtin_amdgcn_wmma_f32_16x16x32_bf16(
            /*neg_a=*/false, a, /*neg_b=*/false, b,
            /*c_mod=*/(short)0, c, /*reuse_a=*/false, /*reuse_b=*/false);
  }
  // C/D layout: VGPR r -> row (r + 8*hi), col = lane%16
  const float bias = b_ih[j0 + l] + b_hh[j0 + l];
  float* zout = Z + (size_t)(t0 + hi * 8) * G4 + j0 + l;
#pragma unroll
  for (int r = 0; r < 8; ++r) zout[(size_t)r * G4] = c[r] + bias;
}

// ---------------- K4: LSTM recurrence via broadcast-A WMMA GEMV ------------
// grid = 2 (one WGP per direction), 512 threads = 16 waves.
// Per step: z[0..1023] = Z[t] + h @ Whh^T computed as 64 16x16 D-tiles,
// A = h replicated across all 16 M rows (lane-group broadcast read of LDS
// bf16 h), so D row M=0 (VGPR 0, lanes 0..15) is the GEMV result.
// Wave w owns N-tiles {4w..4w+3}. Waves 0..7 (j < 512) read B fragments from
// a 256 KB LDS-resident copy of Whh(bf16); waves 8..15 stream from L2 ->
// LDS and L2 bandwidth overlap, halving the per-step streamed bytes.
#define LDS_TILES 32                               // tiles served from LDS
#define BSH_BYTES (LDS_TILES * 8 * 32 * 32)        // 262144
#define ZSH_OFF   0
#define HBF_OFF   (G4 * 4)                         // 4096
#define BSH_OFF   (G4 * 4 + 512)                   // 4608 (32B aligned)
#define LDS_TOTAL (BSH_OFF + BSH_BYTES)            // 266752 < 320 KB

__device__ __forceinline__ float sigmoidf(float x) {
  return 1.0f / (1.0f + __expf(-x));
}

__global__ __launch_bounds__(512) void lstm_recur_wmma(
    const float* __restrict__ Zf, const float* __restrict__ Zb,
    const bf16_t* __restrict__ Whhf, const bf16_t* __restrict__ Whhb,
    const float* __restrict__ h0, const float* __restrict__ c0,
    float* __restrict__ hf, float* __restrict__ hb) {
  extern __shared__ __align__(32) char smem[];
  float*  z_sh = (float*)(smem + ZSH_OFF);     // 1024 f32 gate pre-acts
  bf16_t* hbf  = (bf16_t*)(smem + HBF_OFF);    // 256 bf16 hidden state
  char*   Bsh  = smem + BSH_OFF;               // Whh frags, tiles 0..31

  const int dir = blockIdx.x;                  // 0 = fwd, 1 = bwd
  const float*  Z   = dir ? Zb   : Zf;
  const bf16_t* Whh = dir ? Whhb : Whhf;       // (G4, H2) bf16 row-major
  float*        hs  = dir ? hb   : hf;

  const int tid  = threadIdx.x;
  const int wave = tid >> 5;                   // 0..15
  const int lane = tid & 31;
  const int l    = lane & 15;
  const int hi   = lane >> 4;
  const bool use_lds = (wave < 8);

  // ---- one-time preload: Whh fragments for tiles 0..31 into LDS ----------
  // chunk c = ((ti*8 + k0)*32 + lane): 32B = 16 bf16 per (tile,k0,lane).
  for (int cch = tid; cch < LDS_TILES * 8 * 32; cch += 512) {
    int lanec = cch & 31;
    int k0c   = (cch >> 5) & 7;
    int tic   = cch >> 8;
    int lc = lanec & 15, hic = lanec >> 4;
    const bf16_t* src = Whh + (size_t)(tic * 16 + lc) * H2 + hic * 16 + k0c * 32;
    *(v16bf*)(Bsh + (size_t)cch * 32) = *(const v16bf*)src;
  }

  // ---- init hidden / cell state ------------------------------------------
  if (tid < H2) hbf[tid] = (bf16_t)h0[dir * H2 + tid];
  float c_reg = (tid < H2) ? c0[dir * H2 + tid] : 0.0f;
  __syncthreads();

  for (int t = 0; t < T_SEQ; ++t) {
    const int tt = dir ? (T_SEQ - 1 - t) : t;
    const float* Zrow = Z + (size_t)tt * G4;

    v8f acc[4] = {v8f{}, v8f{}, v8f{}, v8f{}};
#pragma unroll
    for (int k0 = 0; k0 < 8; ++k0) {
      // A fragment: broadcast h -> all M rows identical.
      // lane-group 0: K {32k0..+15}; group 1: +16 elements.
      const char* hp = (const char*)hbf + k0 * 64 + hi * 32;
      v8bf a_lo = *(const v8bf*)(hp);
      v8bf a_hi = *(const v8bf*)(hp + 16);
      v16bf a = __builtin_shufflevector(a_lo, a_hi,
                  0, 1, 2, 3, 4, 5, 6, 7, 8, 9, 10, 11, 12, 13, 14, 15);
#pragma unroll
      for (int i = 0; i < 4; ++i) {
        const int ti = wave * 4 + i;
        v16bf b;
        if (use_lds) {
          b = *(const v16bf*)(Bsh + (size_t)((ti * 8 + k0) * 32 + lane) * 32);
        } else {
          b = *(const v16bf*)(Whh + (size_t)(ti * 16 + l) * H2 + hi * 16 + k0 * 32);
        }
        acc[i] = __builtin_amdgcn_wmma_f32_16x16x32_bf16(
                     false, a, false, b, (short)0, acc[i], false, false);
      }
    }
    // D row M=0 lives in VGPR 0 on lanes 0..15 (N = lane).
#pragma unroll
    for (int i = 0; i < 4; ++i) {
      const int j0i = (wave * 4 + i) * 16;
      if (lane < 16) z_sh[j0i + lane] = acc[i][0] + Zrow[j0i + lane];
    }
    __syncthreads();

    if (tid < H2) {                            // PyTorch gate order i,f,g,o
      float ig = sigmoidf(z_sh[tid]);
      float fg = sigmoidf(z_sh[H2 + tid]);
      float gg = tanhf(z_sh[2 * H2 + tid]);
      float og = sigmoidf(z_sh[3 * H2 + tid]);
      c_reg = fg * c_reg + ig * gg;
      float h = og * tanhf(c_reg);
      hbf[tid] = (bf16_t)h;                    // bf16 operand for next step
      hs[(size_t)tt * H2 + tid] = h;           // f32 trace for feats
    }
    __syncthreads();
  }
}

// ---------------- K5: feats = [hf|hb] @ W_out^T + b_out --------------------
__global__ __launch_bounds__(256) void feats_proj(
    const float* __restrict__ hf, const float* __restrict__ hb,
    const float* __restrict__ Wout, const float* __restrict__ bout,
    float* __restrict__ feats) {
  int idx = blockIdx.x * 256 + threadIdx.x;   // t*TAGS + tag
  if (idx >= T_SEQ * TAGS) return;
  int t = idx / TAGS, tag = idx % TAGS;
  const float* w = Wout + (size_t)tag * (2 * H2);
  const float* a = hf + (size_t)t * H2;
  const float* b = hb + (size_t)t * H2;
  float acc = bout[tag];
  for (int k = 0; k < H2; ++k) acc = fmaf(a[k], w[k], acc);
  for (int k = 0; k < H2; ++k) acc = fmaf(b[k], w[H2 + k], acc);
  feats[idx] = acc;
}

// ---------------- K6: Viterbi scan + backtrack (single block) ---------------
__global__ __launch_bounds__(64) void viterbi(
    const float* __restrict__ feats, const float* __restrict__ trans,
    int* __restrict__ bp, float* __restrict__ out) {
  __shared__ float fv[TAGS];
  __shared__ float tr[TAGS * TAGS];
  const int j = threadIdx.x;
  if (j < TAGS * TAGS) tr[j] = trans[j];
  if (j < TAGS) fv[j] = (j == TSTART) ? 0.0f : NEGV;
  __syncthreads();

  for (int t = 0; t < T_SEQ; ++t) {
    float best = 0.0f; int arg = 0;
    if (j < TAGS) {                           // tmp[j,p] = trans[j,p] + fv[p]
      best = tr[j * TAGS] + fv[0];
      for (int p = 1; p < TAGS; ++p) {
        float v = tr[j * TAGS + p] + fv[p];
        if (v > best) { best = v; arg = p; }  // first-max tie-break (argmax)
      }
      bp[t * TAGS + j] = arg;
    }
    __syncthreads();
    if (j < TAGS) fv[j] = best + feats[t * TAGS + j];
    __syncthreads();
  }

  if (j == 0) {
    float bestv = fv[0] + tr[TSTOP * TAGS + 0]; int bt = 0;
    for (int p = 1; p < TAGS; ++p) {
      float v = fv[p] + tr[TSTOP * TAGS + p];
      if (v > bestv) { bestv = v; bt = p; }
    }
    out[0] = bestv;
    int tag = bt;
    for (int t = T_SEQ - 1; t >= 0; --t) {    // reverse scan: path[T-1]=best
      out[1 + t] = (float)tag;
      tag = bp[t * TAGS + tag];
    }
  }
}

// ---------------------------------------------------------------------------
extern "C" void kernel_launch(void* const* d_in, const int* in_sizes, int n_in,
                              void* d_out, int out_size, void* d_ws, size_t ws_size,
                              hipStream_t stream) {
  const int*   sentence = (const int*)  d_in[0];
  const float* emb      = (const float*)d_in[1];
  const float* w_ih_f   = (const float*)d_in[2];
  const float* w_hh_f   = (const float*)d_in[3];
  const float* b_ih_f   = (const float*)d_in[4];
  const float* b_hh_f   = (const float*)d_in[5];
  const float* w_ih_b   = (const float*)d_in[6];
  const float* w_hh_b   = (const float*)d_in[7];
  const float* b_ih_b   = (const float*)d_in[8];
  const float* b_hh_b   = (const float*)d_in[9];
  const float* W_out    = (const float*)d_in[10];
  const float* b_out    = (const float*)d_in[11];
  const float* trans    = (const float*)d_in[12];
  const float* h0       = (const float*)d_in[13];
  const float* c0       = (const float*)d_in[14];
  float* out = (float*)d_out;

  // ---- workspace carve-up (256-byte aligned slabs) ----
  char* ws = (char*)d_ws;
  size_t o = 0;
  bf16_t* Xbf  = (bf16_t*)(ws + o); o += (size_t)T_SEQ * EMB * 2;       // 2 MB
  bf16_t* Wfb  = (bf16_t*)(ws + o); o += (size_t)G4 * EMB * 2;          // 512 KB
  bf16_t* Wbb  = (bf16_t*)(ws + o); o += (size_t)G4 * EMB * 2;          // 512 KB
  bf16_t* Whfb = (bf16_t*)(ws + o); o += (size_t)G4 * H2 * 2;           // 512 KB
  bf16_t* Whbb = (bf16_t*)(ws + o); o += (size_t)G4 * H2 * 2;           // 512 KB
  float*  Zf   = (float*)(ws + o);  o += (size_t)T_SEQ * G4 * 4;        // 16 MB
  float*  Zb   = (float*)(ws + o);  o += (size_t)T_SEQ * G4 * 4;        // 16 MB
  float*  hf   = (float*)(ws + o);  o += (size_t)T_SEQ * H2 * 4;        // 4 MB
  float*  hb   = (float*)(ws + o);  o += (size_t)T_SEQ * H2 * 4;        // 4 MB
  float*  ft   = (float*)(ws + o);  o += (size_t)T_SEQ * TAGS * 4;
  int*    bp   = (int*)(ws + o);    o += (size_t)T_SEQ * TAGS * 4;
  (void)ws_size; (void)n_in; (void)in_sizes; (void)out_size;

  // K1: gather + convert embeddings (T*EMB/4 quads)
  gather_embed_bf16<<<(T_SEQ * (EMB / 4) + 255) / 256, 256, 0, stream>>>(
      sentence, emb, Xbf);

  // K2: convert weights to bf16
  const int nW = G4 * EMB;
  cvt_bf16<<<(nW + 255) / 256, 256, 0, stream>>>(w_ih_f, Wfb, nW);
  cvt_bf16<<<(nW + 255) / 256, 256, 0, stream>>>(w_ih_b, Wbb, nW);
  cvt_bf16<<<(nW + 255) / 256, 256, 0, stream>>>(w_hh_f, Whfb, nW);
  cvt_bf16<<<(nW + 255) / 256, 256, 0, stream>>>(w_hh_b, Whbb, nW);

  // K3: WMMA input projections, 256x64 tiles, 8 tiles (waves) per block
  const int nTiles = (T_SEQ / 16) * (G4 / 16);
  gemm_z_wmma<<<nTiles / 8, 256, 0, stream>>>(Xbf, Wfb, b_ih_f, b_hh_f, Zf);
  gemm_z_wmma<<<nTiles / 8, 256, 0, stream>>>(Xbf, Wbb, b_ih_b, b_hh_b, Zb);

  // K4: both directions' recurrences concurrently (grid = 2, 1 WGP each,
  //     266752 B dynamic LDS: z_sh + hbf + 256 KB Whh fragment cache)
  lstm_recur_wmma<<<2, 512, LDS_TOTAL, stream>>>(
      Zf, Zb, Whfb, Whbb, h0, c0, hf, hb);

  // K5: tag scores
  feats_proj<<<(T_SEQ * TAGS + 255) / 256, 256, 0, stream>>>(
      hf, hb, W_out, b_out, ft);

  // K6: Viterbi decode -> out[0]=score, out[1..T]=path
  viterbi<<<1, 64, 0, stream>>>(ft, trans, bp, out);
}
